// SAModule_8229157339244
// MI455X (gfx1250) — matrix operand order repaired and checked by reading, not compiled, not run on hardware
//
#include <hip/hip_runtime.h>
#include <hip/hip_bf16.h>

// ---------------------------------------------------------------------------
// PointNet++ SA module on MI455X (gfx1250, wave32).
// Per-edge MLP as bf16 WMMA GEMM (v_wmma_f32_16x16x32_bf16), segment-max via
// integer atomic max on non-negative float bit patterns.
// Block = 16 edges x 128 hidden, 8 waves; each wave owns a 16-wide N tile and
// runs 5 WMMA k-steps (K = 132 padded to 160).
// ---------------------------------------------------------------------------

typedef __attribute__((ext_vector_type(16))) __bf16 v16bf;
typedef __attribute__((ext_vector_type(8)))  __bf16 v8bf;
typedef __attribute__((ext_vector_type(8)))  float  v8f;

#define FDIM 64          // features per node
#define HDIM 128         // hidden width
#define KDIM 132         // 2F + 4
#define KPAD 160         // 5 * 32
#define ASTRIDE 168      // LDS row stride (bf16 elems), 16B-aligned rows

// ---- f32 -> bf16 copy of x ------------------------------------------------
__global__ __launch_bounds__(256) void convert_x_kernel(
    const float* __restrict__ x, __bf16* __restrict__ xb, int n) {
  int t = blockIdx.x * 256 + threadIdx.x;
  if (t < n) xb[t] = (__bf16)x[t];
}

// ---- W [132,128] f32 -> Wt [128 cols][160 k] bf16, zero-padded -------------
__global__ __launch_bounds__(256) void prep_w_kernel(
    const float* __restrict__ W, __bf16* __restrict__ Wt) {
  int t = blockIdx.x * 256 + threadIdx.x;
  if (t >= KPAD * HDIM) return;
  int n = t / KPAD;
  int k = t % KPAD;
  Wt[t] = (k < KDIM) ? (__bf16)W[k * HDIM + n] : (__bf16)0.0f;
}

__device__ __forceinline__ float dev_angle(float ax, float ay, float az,
                                           float bx, float by, float bz) {
  // atan2(||a x b||, a . b)
  float cx = ay * bz - az * by;
  float cy = az * bx - ax * bz;
  float cz = ax * by - ay * bx;
  float cn = sqrtf(cx * cx + cy * cy + cz * cz);
  float d  = ax * bx + ay * by + az * bz;
  return atan2f(cn, d);
}

// ---- main fused kernel: gather + ppf + bf16 WMMA GEMM + relu + atomic max --
__global__ __launch_bounds__(256) void edge_gemm_kernel(
    const __bf16* __restrict__ xb,      // [N, 64] bf16
    const __bf16* __restrict__ Wt,      // [128, 160] bf16 (col-major K)
    const float*  __restrict__ bias,    // [128]
    const int*    __restrict__ e0arr,   // [E] sources
    const int*    __restrict__ e1arr,   // [E] targets
    const float*  __restrict__ pos,     // [N,3]
    const float*  __restrict__ normal,  // [N,3]
    float*        __restrict__ agg,     // [N,128], zero-initialized
    int E) {
  __shared__ __bf16 sA[16 * ASTRIDE];
  __shared__ int sE0[16];
  __shared__ int sE1[16];

  const int tid  = threadIdx.x;
  const int base = blockIdx.x * 16;

  if (tid < 16) {
    int e = base + tid;
    if (e >= E) e = E - 1;
    sE0[tid] = e0arr[e];
    sE1[tid] = e1arr[e];
  }
  __syncthreads();

  // Cooperative A-tile fill: cols [0,128) from xb, [132,160) zero.
  for (int i = tid; i < 16 * KPAD; i += 256) {
    int r = i / KPAD;
    int k = i % KPAD;
    if (k >= 2 * FDIM && k < KDIM) continue;  // ppf slots filled below
    __bf16 v = (__bf16)0.0f;
    if (k < 2 * FDIM) {
      int node = (k < FDIM) ? sE1[r] : sE0[r];  // cat[x[e1], x[e0], ppf]
      int kk   = (k < FDIM) ? k : (k - FDIM);
      v = xb[(size_t)node * FDIM + kk];
    }
    sA[r * ASTRIDE + k] = v;
  }

  // Point-pair features for the 16 edges of this tile (lanes 0..15).
  if (tid < 16) {
    int a = sE0[tid], c = sE1[tid];
    float px = pos[a * 3 + 0] - pos[c * 3 + 0];
    float py = pos[a * 3 + 1] - pos[c * 3 + 1];
    float pz = pos[a * 3 + 2] - pos[c * 3 + 2];
    float n0x = normal[a * 3 + 0], n0y = normal[a * 3 + 1], n0z = normal[a * 3 + 2];
    float n1x = normal[c * 3 + 0], n1y = normal[c * 3 + 1], n1z = normal[c * 3 + 2];
    float p0 = sqrtf(px * px + py * py + pz * pz);
    float p1 = dev_angle(n1x, n1y, n1z, px, py, pz);
    float p2 = dev_angle(n0x, n0y, n0z, px, py, pz);
    float p3 = dev_angle(n1x, n1y, n1z, n0x, n0y, n0z);
    __bf16* row = &sA[tid * ASTRIDE + 2 * FDIM];
    row[0] = (__bf16)p0;
    row[1] = (__bf16)p1;
    row[2] = (__bf16)p2;
    row[3] = (__bf16)p3;
  }
  __syncthreads();

  // WMMA: each wave computes a 16x16 f32 tile over 5 k-steps of 32.
  const int wave = tid >> 5;
  const int lane = tid & 31;
  const int m    = lane & 15;
  const int hi   = lane >> 4;          // lane half selector
  const int n    = wave * 16 + m;      // output column owned by this lane

  v8f acc = {};
#pragma unroll
  for (int t = 0; t < 5; ++t) {
    // A fragment (16-bit A 16x32 layout): two contiguous 16B chunks per lane.
    const __bf16* ap = &sA[m * ASTRIDE + t * 32 + 8 * hi];
    v8bf alo = *(const v8bf*)ap;          // k = t*32 + 8*hi + [0..7]
    v8bf ahi = *(const v8bf*)(ap + 16);   // k = t*32 + 8*hi + 16 + [0..7]
    // B fragment (16-bit B 32x16 layout): two contiguous 16B chunks per lane.
    const __bf16* bp = &Wt[(size_t)n * KPAD + t * 32 + 8 * hi];
    v8bf blo = *(const v8bf*)bp;          // k = t*32 + 8*hi + [0..7]
    v8bf bhi = *(const v8bf*)(bp + 16);   // k = t*32 + 8*hi + 16 + [0..7]
    v16bf A, B;
#pragma unroll
    for (int e2 = 0; e2 < 8; ++e2) {
      A[e2]     = alo[e2];
      A[e2 + 8] = ahi[e2];
      B[e2]     = blo[e2];
      B[e2 + 8] = bhi[e2];
    }
    acc = __builtin_amdgcn_wmma_f32_16x16x32_bf16(
        /*neg_a=*/false, A, /*neg_b=*/false, B,
        /*c_mod=*/(short)0, acc, /*reuse_a=*/false, /*reuse_b=*/false);
  }

  // Epilogue: bias + relu + segment-max via int atomicMax (vals >= 0).
  float bn = bias[n];
#pragma unroll
  for (int v = 0; v < 8; ++v) {
    int row  = v + (hi << 3);           // C/D layout: M = v + 8*hi
    int edge = base + row;
    if (edge < E) {
      float val = fmaxf(acc[v] + bn, 0.0f);
      int tgt = sE1[row];
      atomicMax((int*)&agg[(size_t)tgt * HDIM + n], __float_as_int(val));
    }
  }
}

// ---- gather: out[S,128] = agg[idx], out2[S,3] = pos[idx] -------------------
__global__ __launch_bounds__(256) void gather_out_kernel(
    const float* __restrict__ agg, const float* __restrict__ pos,
    const int* __restrict__ idx, float* __restrict__ out, int S) {
  int t = blockIdx.x * 256 + threadIdx.x;
  int total = S * HDIM + S * 3;
  if (t >= total) return;
  if (t < S * HDIM) {
    int s = t / HDIM, h = t % HDIM;
    out[t] = agg[(size_t)idx[s] * HDIM + h];
  } else {
    int t2 = t - S * HDIM;
    int s = t2 / 3, c = t2 % 3;
    out[t] = pos[(size_t)idx[s] * 3 + c];
  }
}

extern "C" void kernel_launch(void* const* d_in, const int* in_sizes, int n_in,
                              void* d_out, int out_size, void* d_ws, size_t ws_size,
                              hipStream_t stream) {
  const float* x        = (const float*)d_in[0];
  const float* pos      = (const float*)d_in[1];
  const float* normal   = (const float*)d_in[2];
  const float* W        = (const float*)d_in[3];
  const float* b        = (const float*)d_in[4];
  const int*   edge_idx = (const int*)d_in[5];
  const int*   idx      = (const int*)d_in[6];

  const int N = in_sizes[0] / FDIM;
  const int E = in_sizes[5] / 2;
  const int S = in_sizes[6];

  const int* e0 = edge_idx;       // sources (col)
  const int* e1 = edge_idx + E;   // targets (row)

  // Workspace layout (256B-aligned sections).
  char* ws = (char*)d_ws;
  size_t aggB = ((size_t)N * HDIM * sizeof(float) + 255) & ~(size_t)255;
  size_t xbB  = ((size_t)N * FDIM * sizeof(__bf16) + 255) & ~(size_t)255;
  float*  agg = (float*)ws;
  __bf16* xb  = (__bf16*)(ws + aggB);
  __bf16* Wt  = (__bf16*)(ws + aggB + xbB);

  hipMemsetAsync(agg, 0, (size_t)N * HDIM * sizeof(float), stream);

  convert_x_kernel<<<(N * FDIM + 255) / 256, 256, 0, stream>>>(x, xb, N * FDIM);
  prep_w_kernel<<<(KPAD * HDIM + 255) / 256, 256, 0, stream>>>(W, Wt);
  edge_gemm_kernel<<<(E + 15) / 16, 256, 0, stream>>>(
      xb, Wt, b, e0, e1, pos, normal, agg, E);
  gather_out_kernel<<<(S * (HDIM + 3) + 255) / 256, 256, 0, stream>>>(
      agg, pos, idx, (float*)d_out, S);
}
